// MultiHeadAttention_3633542332805
// MI455X (gfx1250) — matrix operand order repaired and checked by reading, not compiled, not run on hardware
//
#include <hip/hip_runtime.h>

typedef _Float16 half_t;
typedef __attribute__((ext_vector_type(16))) _Float16 v16h;
typedef __attribute__((ext_vector_type(8)))  _Float16 v8h;
typedef __attribute__((ext_vector_type(4)))  _Float16 v4h;
typedef __attribute__((ext_vector_type(8)))  float    v8f;

#define S_LEN 4096
#define EMBD  1024
#define NHEAD 16
#define HDIM  64

// 1/sqrt(N_EMBD) * log2(e): folded into Q so scores are in exp2 domain.
#define QSCALE 0.04508422f

// hardware exp2 (v_exp_f32)
static __device__ __forceinline__ float fast_exp2(float x) {
  return __builtin_amdgcn_exp2f(x);
}

// ---------------------------------------------------------------------------
// WMMA wrapper: D = A(16x32 f16) * B(32x16 f16) + C(16x16 f32)
// ---------------------------------------------------------------------------
static __device__ __forceinline__ v8f wmma_f16(v16h a, v16h b, v8f c) {
  return __builtin_amdgcn_wmma_f32_16x16x32_f16(
      /*neg_a=*/false, a, /*neg_b=*/false, b,
      /*c_mod=*/(short)0, c, /*reuse_a=*/false, /*reuse_b=*/false);
}

// ---------------------------------------------------------------------------
// gfx1250 async global->LDS copy (ASYNCcnt-tracked), GVS addressing:
//   LDS[vdst] = MEM[saddr64 + voffset]   (16 bytes per lane)
// Generic LDS pointer low 32 bits == LDS byte address (ISA aperture rule).
// ---------------------------------------------------------------------------
static __device__ __forceinline__ void async_load_b128(const void* gbase,
                                                       unsigned goff,
                                                       void* lds_ptr) {
  const unsigned lds = (unsigned)(size_t)lds_ptr;
  asm volatile("global_load_async_to_lds_b128 %0, %1, %2"
               :
               : "v"(lds), "v"(goff), "s"((unsigned long long)(size_t)gbase)
               : "memory");
}
static __device__ __forceinline__ void wait_async() {
  asm volatile("s_wait_asynccnt 0" ::: "memory");
}

// A-fragment (16x32, row-major source, row stride ld halves).
// ISA 7.12.2: lane m = lane&15; lane half selects K block 0-7/16-23 vs 8-15/24-31.
// Each half of the fragment is 8 contiguous halves -> 2x ds_load_b128.
static __device__ __forceinline__ v16h load_A_frag(const half_t* tile, int ld) {
  const int lane = threadIdx.x & 31;
  const int m = lane & 15;
  const int kh = (lane >> 4) << 3;  // 0 or 8
  const half_t* row = tile + m * ld;
  v16h f;
#pragma unroll
  for (int r = 0; r < 8; ++r) {
    const int k0 = (r < 4) ? (kh + 2 * r) : (16 + kh + 2 * (r - 4));
    f[2 * r]     = row[k0];
    f[2 * r + 1] = row[k0 + 1];
  }
  return f;
}

// B-fragment from a transposed source: tile stored [n][k] (row stride ld).
// lane n = lane&15; lanes 0-15 hold K=0..7,16..23, lanes 16-31 K=8..15,24..31.
// two runs of 8 contiguous halves per lane -> 2x ds_load_b128.
static __device__ __forceinline__ v16h load_Bt_frag(const half_t* tile, int ld) {
  const int lane = threadIdx.x & 31;
  const int n = lane & 15;
  const int kb = (lane >> 4) << 3;
  const half_t* row = tile + n * ld + kb;
  v16h f;
#pragma unroll
  for (int r = 0; r < 16; ++r) f[r] = row[r < 8 ? r : r + 8];
  return f;
}

// ---------------------------------------------------------------------------
// f32 -> f16 conversion (4 elements / thread)
// ---------------------------------------------------------------------------
__global__ void cvt_f32_f16_kernel(const float* __restrict__ s,
                                   half_t* __restrict__ d, int n4) {
  const int i = blockIdx.x * 256 + threadIdx.x;
  if (i < n4) {
    const float4 f = ((const float4*)s)[i];
    v4h o;
    o[0] = (half_t)f.x; o[1] = (half_t)f.y;
    o[2] = (half_t)f.z; o[3] = (half_t)f.w;
    ((v4h*)d)[i] = o;
  }
}

// ---------------------------------------------------------------------------
// Tiled WMMA GEMM: C[M,N] = A[M,K](f16) * B[K,N](f16) + bias
// EPI==0: scatter into Q/K/V f16 buffers laid out [head][s][d]
//         (Q is pre-scaled by QSCALE so attention scores land in exp2 domain)
// EPI==1: write f32 to outF
// BM=128, BN=64, BK=64; 256 threads (8 waves); each wave owns a 32x32 block.
// A tiles: async DMA to LDS (double-buffered). B tiles: staged transposed.
// ---------------------------------------------------------------------------
template <int EPI>
__global__ __launch_bounds__(256) void wmma_gemm_kernel(
    const half_t* __restrict__ A, const half_t* __restrict__ B,
    const float* __restrict__ bias,
    half_t* __restrict__ qd, half_t* __restrict__ kd, half_t* __restrict__ vd,
    float* __restrict__ outF, int M, int N, int K) {
  constexpr int BM = 128, BN = 64, BK = 64;
  constexpr int LDA = BK + 8, LDB = BK + 8;  // Bs is [n][k]
  __shared__ __align__(16) half_t As[2][BM * LDA];
  __shared__ __align__(16) half_t Bs[2][BN * LDB];

  const int tid  = threadIdx.x;
  const int wave = tid >> 5;
  const int wr = wave >> 1;  // 0..3 -> M sub-block
  const int wc = wave & 1;   // 0..1 -> N sub-block
  const int lane = tid & 31;
  const int mBase = blockIdx.y * BM;
  const int nBase = blockIdx.x * BN;

  const v8f vzero = {};
  v8f acc[2][2];
#pragma unroll
  for (int i = 0; i < 2; ++i)
#pragma unroll
    for (int j = 0; j < 2; ++j) acc[i][j] = vzero;

  auto stage = [&](int ko, int buf) {
    // A tile 128x64 halves: 4 async 16B chunks per thread
#pragma unroll
    for (int i = 0; i < (BM * BK) / (8 * 256); ++i) {
      const int c = tid + i * 256;
      const int row = c >> 3;
      const int col = (c & 7) * 8;
      async_load_b128(A, (unsigned)(((size_t)(mBase + row) * K + ko + col) * 2),
                      &As[buf][row * LDA + col]);
    }
    // B tile 64x64 halves, store transposed [n][k]
#pragma unroll
    for (int i = 0; i < (BK * BN) / (8 * 256); ++i) {
      const int c = tid + i * 256;
      const int krow = c >> 3;
      const int col = (c & 7) * 8;
      const v8h bv = *(const v8h*)&B[(size_t)(ko + krow) * N + nBase + col];
#pragma unroll
      for (int j = 0; j < 8; ++j) Bs[buf][(col + j) * LDB + krow] = bv[j];
    }
  };

  stage(0, 0);
  const int iters = K / BK;
  for (int it = 0; it < iters; ++it) {
    wait_async();
    __syncthreads();
    if (it + 1 < iters) stage((it + 1) * BK, (it + 1) & 1);

    const half_t* Asb = &As[it & 1][0];
    const half_t* Bsb = &Bs[it & 1][0];
#pragma unroll
    for (int kk = 0; kk < BK; kk += 32) {
      v16h a0 = load_A_frag(&Asb[(wr * 32 + 0)  * LDA + kk], LDA);
      v16h a1 = load_A_frag(&Asb[(wr * 32 + 16) * LDA + kk], LDA);
      v16h b0 = load_Bt_frag(&Bsb[(wc * 32 + 0)  * LDB + kk], LDB);
      v16h b1 = load_Bt_frag(&Bsb[(wc * 32 + 16) * LDB + kk], LDB);
      acc[0][0] = wmma_f16(a0, b0, acc[0][0]);
      acc[0][1] = wmma_f16(a0, b1, acc[0][1]);
      acc[1][0] = wmma_f16(a1, b0, acc[1][0]);
      acc[1][1] = wmma_f16(a1, b1, acc[1][1]);
    }
  }

  // Epilogue. C layout: col = lane&15, row = r + 8*(lane>>4) per VGPR r.
  const int cn   = lane & 15;
  const int rOff = (lane >> 4) * 8;
#pragma unroll
  for (int ti = 0; ti < 2; ++ti) {
#pragma unroll
    for (int tj = 0; tj < 2; ++tj) {
      const int tileRow = mBase + wr * 32 + ti * 16;
      const int tileCol = nBase + wc * 32 + tj * 16;
      const int col = tileCol + cn;
      const float bv = bias[col];
#pragma unroll
      for (int r = 0; r < 8; ++r) {
        const int row = tileRow + rOff + r;
        float val = acc[ti][tj][r] + bv;
        if (EPI == 0) {
          // qkv column c = h*192 + which*64 + d
          const int h     = col / 192;
          const int rem   = col - h * 192;
          const int which = rem >> 6;
          const int dcol  = rem & 63;
          if (which == 0) val *= QSCALE;  // pre-scale Q
          half_t* dst = (which == 0) ? qd : (which == 1) ? kd : vd;
          dst[((size_t)h * S_LEN + row) * HDIM + dcol] = (half_t)val;
        } else {
          outF[(size_t)row * N + col] = val;
        }
      }
    }
  }
}

// ---------------------------------------------------------------------------
// Flash attention (causal). Grid: (S/128, NHEAD). 256 threads = 8 waves;
// each wave owns 16 query rows. K tiles async-DMA'd to LDS, V tiles staged
// transposed [d][key]; both double-buffered. Softmax in exp2 domain; row
// sums computed with WMMA against a ones matrix.
// ---------------------------------------------------------------------------
__global__ __launch_bounds__(256) void flash_attn_kernel(
    const half_t* __restrict__ Qg, const half_t* __restrict__ Kg,
    const half_t* __restrict__ Vg, half_t* __restrict__ Ctx) {
  constexpr int LDT = HDIM + 8;  // 72
  __shared__ __align__(16) half_t Ks[2][64 * LDT];  // [key][d]
  __shared__ __align__(16) half_t Vt[2][64 * LDT];  // [d][key]
  __shared__ __align__(16) half_t Ps[8 * 16 * LDT]; // per-wave P bounce

  const int tid  = threadIdx.x;
  const int wave = tid >> 5;
  const int lane = tid & 31;
  const int h = blockIdx.y;
  const int qBase = blockIdx.x * 128;
  const int qRow  = qBase + wave * 16;

  const half_t* Qh = Qg + (size_t)h * S_LEN * HDIM;
  const half_t* Kh = Kg + (size_t)h * S_LEN * HDIM;
  const half_t* Vh = Vg + (size_t)h * S_LEN * HDIM;

  // resident Q fragments for this wave's 16 rows (d chunks 0-31, 32-63)
  const v16h qf0 = load_A_frag(Qh + (size_t)qRow * HDIM, HDIM);
  const v16h qf1 = load_A_frag(Qh + (size_t)qRow * HDIM + 32, HDIM);

  const v8f vzero = {};
  v16h ones;
#pragma unroll
  for (int i = 0; i < 16; ++i) ones[i] = (half_t)1.0f;

  float mrow[8], lrow[8];
  v8f oacc[4];
#pragma unroll
  for (int r = 0; r < 8; ++r) { mrow[r] = -1.0e30f; lrow[r] = 0.0f; }
#pragma unroll
  for (int dt = 0; dt < 4; ++dt) oacc[dt] = vzero;

  const int cn   = lane & 15;
  const int rOff = (lane >> 4) * 8;

  auto stage = [&](int jt, int buf) {
    const int kB = jt * 64;
    // K tile 64x64 halves: 2 async 16B chunks per thread
#pragma unroll
    for (int i = 0; i < 2; ++i) {
      const int c = tid + i * 256;
      const int row = c >> 3, col = (c & 7) * 8;
      async_load_b128(Kh, (unsigned)(((size_t)(kB + row) * HDIM + col) * 2),
                      &Ks[buf][row * LDT + col]);
    }
    // V tile, store transposed [d][key]
#pragma unroll
    for (int i = 0; i < 2; ++i) {
      const int c = tid + i * 256;
      const int row = c >> 3, col = (c & 7) * 8;
      const v8h vv = *(const v8h*)&Vh[(size_t)(kB + row) * HDIM + col];
#pragma unroll
      for (int j = 0; j < 8; ++j) Vt[buf][(col + j) * LDT + row] = vv[j];
    }
  };

  const int jtEnd = (qBase + 128) / 64;
  stage(0, 0);
  for (int jt = 0; jt < jtEnd; ++jt) {
    const int kBase = jt * 64;
    wait_async();
    __syncthreads();
    if (jt + 1 < jtEnd) stage(jt + 1, (jt + 1) & 1);

    if (kBase <= qRow + 15) {  // wave-uniform: at least one unmasked key
      const half_t* Kst = &Ks[jt & 1][0];
      const half_t* Vst = &Vt[jt & 1][0];
      // Z = Qscaled * K^T for 16 rows x 64 keys (already in exp2 domain)
      v8f sf[4];
#pragma unroll
      for (int nt = 0; nt < 4; ++nt) {
        v8f s = vzero;
        v16h b0 = load_Bt_frag(&Kst[(nt * 16) * LDT + 0],  LDT);
        v16h b1 = load_Bt_frag(&Kst[(nt * 16) * LDT + 32], LDT);
        s = wmma_f16(qf0, b0, s);
        s = wmma_f16(qf1, b1, s);
        sf[nt] = s;
      }
      // causal mask: only needed on tiles overlapping the diagonal
      const bool needMask = (kBase + 63 > qRow);
      if (needMask) {
#pragma unroll
        for (int nt = 0; nt < 4; ++nt) {
          const int col = kBase + nt * 16 + cn;
#pragma unroll
          for (int r = 0; r < 8; ++r)
            if (col > qRow + rOff + r) sf[nt][r] = -1.0e30f;
        }
      }
      // per-row running max (reductions stay within each 16-lane half)
      float mnew[8];
#pragma unroll
      for (int r = 0; r < 8; ++r)
        mnew[r] = fmaxf(fmaxf(sf[0][r], sf[1][r]),
                        fmaxf(sf[2][r], sf[3][r]));
#pragma unroll
      for (int r = 0; r < 8; ++r) {
#pragma unroll
        for (int x = 1; x < 16; x <<= 1)
          mnew[r] = fmaxf(mnew[r], __shfl_xor(mnew[r], x, 32));
        mnew[r] = fmaxf(mnew[r], mrow[r]);
      }
      // p = 2^(z - m); rescale running state
      float alpha[8];
#pragma unroll
      for (int r = 0; r < 8; ++r) {
        alpha[r] = fast_exp2(mrow[r] - mnew[r]);
        mrow[r] = mnew[r];
#pragma unroll
        for (int dt = 0; dt < 4; ++dt) oacc[dt][r] *= alpha[r];
      }
      // bounce P (16x64 f32 C-layout) through wave-private LDS as f16,
      // then reload as A fragments. DS ops are in-order within a wave.
      half_t* P = &Ps[wave * 16 * LDT];
#pragma unroll
      for (int nt = 0; nt < 4; ++nt)
#pragma unroll
        for (int r = 0; r < 8; ++r)
          P[(rOff + r) * LDT + nt * 16 + cn] =
              (half_t)fast_exp2(sf[nt][r] - mnew[r]);
      const v16h pa0 = load_A_frag(P, LDT);       // keys 0..31
      const v16h pa1 = load_A_frag(P + 32, LDT);  // keys 32..63
      // row sums of P via WMMA against ones: every column of C = rowsum,
      // landing exactly in the per-lane row-stat layout.
      v8f rsum = vzero;
      rsum = wmma_f16(pa0, ones, rsum);
      rsum = wmma_f16(pa1, ones, rsum);
#pragma unroll
      for (int r = 0; r < 8; ++r) lrow[r] = lrow[r] * alpha[r] + rsum[r];
      // O += P * V   (B frags from transposed V: contiguous per lane)
#pragma unroll
      for (int dt = 0; dt < 4; ++dt) {
        v16h vb0 = load_Bt_frag(&Vst[(dt * 16) * LDT + 0],  LDT);
        v16h vb1 = load_Bt_frag(&Vst[(dt * 16) * LDT + 32], LDT);
        oacc[dt] = wmma_f16(pa0, vb0, oacc[dt]);
        oacc[dt] = wmma_f16(pa1, vb1, oacc[dt]);
      }
    }
  }

  // normalize and write ctx (f16, row-major [S][EMBD])
#pragma unroll
  for (int r = 0; r < 8; ++r) {
    const int row = qRow + rOff + r;
    const float inv = 1.0f / lrow[r];
#pragma unroll
    for (int dt = 0; dt < 4; ++dt)
      Ctx[(size_t)row * EMBD + h * HDIM + dt * 16 + cn] =
          (half_t)(oacc[dt][r] * inv);
  }
}

// ---------------------------------------------------------------------------
// Host launcher
// ---------------------------------------------------------------------------
extern "C" void kernel_launch(void* const* d_in, const int* in_sizes, int n_in,
                              void* d_out, int out_size, void* d_ws,
                              size_t ws_size, hipStream_t stream) {
  (void)in_sizes; (void)n_in; (void)out_size; (void)ws_size;
  const float* X    = (const float*)d_in[0];  // [4096,1024]
  const float* Wqkv = (const float*)d_in[1];  // [1024,3072]
  const float* Bqkv = (const float*)d_in[2];  // [3072]
  const float* Wout = (const float*)d_in[3];  // [1024,1024]
  const float* Bout = (const float*)d_in[4];  // [1024]
  float* out = (float*)d_out;                 // [4096,1024] f32

  char* ws = (char*)d_ws;
  const size_t MiB = (size_t)1 << 20;
  half_t* Xh   = (half_t*)(ws + 0);         //  8 MiB
  half_t* Wqh  = (half_t*)(ws + 8 * MiB);   //  6 MiB
  half_t* Woh  = (half_t*)(ws + 14 * MiB);  //  2 MiB
  half_t* Qh   = (half_t*)(ws + 16 * MiB);  //  8 MiB  [head][s][d]
  half_t* Kh   = (half_t*)(ws + 24 * MiB);  //  8 MiB
  half_t* Vh   = (half_t*)(ws + 32 * MiB);  //  8 MiB
  half_t* Ctxh = (half_t*)(ws + 40 * MiB);  //  8 MiB  [s][embd]

  {
    int n4 = (S_LEN * EMBD) / 4;
    cvt_f32_f16_kernel<<<(n4 + 255) / 256, 256, 0, stream>>>(X, Xh, n4);
    n4 = (EMBD * 3 * EMBD) / 4;
    cvt_f32_f16_kernel<<<(n4 + 255) / 256, 256, 0, stream>>>(Wqkv, Wqh, n4);
    n4 = (EMBD * EMBD) / 4;
    cvt_f32_f16_kernel<<<(n4 + 255) / 256, 256, 0, stream>>>(Wout, Woh, n4);
  }

  // QKV projection -> Q/K/V f16 [head][s][d]  (Q pre-scaled)
  wmma_gemm_kernel<0><<<dim3(3 * EMBD / 64, S_LEN / 128), 256, 0, stream>>>(
      Xh, Wqh, Bqkv, Qh, Kh, Vh, nullptr, S_LEN, 3 * EMBD, EMBD);

  // causal flash attention -> ctx f16 [s][embd]
  flash_attn_kernel<<<dim3(S_LEN / 128, NHEAD), 256, 0, stream>>>(Qh, Kh, Vh,
                                                                  Ctxh);

  // output projection -> f32 out
  wmma_gemm_kernel<1><<<dim3(EMBD / 64, S_LEN / 128), 256, 0, stream>>>(
      Ctxh, Woh, Bout, nullptr, nullptr, nullptr, out, S_LEN, EMBD, EMBD);
}
